// SelfAttention_81037442941296
// MI455X (gfx1250) — matrix operand order repaired and checked by reading, not compiled, not run on hardware
//
#include <hip/hip_runtime.h>
#include <hip/hip_bf16.h>

// ---------------------------------------------------------------------------
// Self-attention (n=4096, d_model=1024) for MI455X / gfx1250.
// All heavy math runs on v_wmma_f32_16x16x32_bf16 (f32 accumulate).
// Workspace requirement: 80 MB.
// ---------------------------------------------------------------------------

#define N_TOK 4096
#define D_IN  1024
#define D_OUT 1024
#define Q_SCALE 0.06f

typedef __attribute__((ext_vector_type(16))) __bf16 v16bf;
typedef __attribute__((ext_vector_type(8)))  __bf16 v8bf;
typedef __attribute__((ext_vector_type(8)))  float  v8f;

// ---------------- fragment loaders (layouts per CDNA5 ISA 7.12.2) ----------

// A-matrix 16x32 bf16, row-major with leading dim lda (elements).
// lane l (0..31): row m = l&15 ; K-base = (l>>4)*8
//   vector elems 0..7  = A[m][k0+kb .. k0+kb+7]
//   vector elems 8..15 = A[m][k0+kb+16 .. k0+kb+23]
__device__ __forceinline__ v16bf load_a_frag(const __bf16* __restrict__ A,
                                             int lda, int row0, int k0, int lane) {
    const int m  = lane & 15;
    const int kb = (lane >> 4) * 8;
    const __bf16* p = A + (size_t)(row0 + m) * (size_t)lda + (size_t)(k0 + kb);
    v8bf lo = *(const v8bf*)(p);
    v8bf hi = *(const v8bf*)(p + 16);
    v16bf a;
#pragma unroll
    for (int i = 0; i < 8; ++i) { a[i] = lo[i]; a[8 + i] = hi[i]; }
    return a;
}

// B-matrix 32x16 bf16 taken from Bt = B^T, row-major [N x K], ldbt = K.
// lane l: column c = l&15 ; K-base = (l>>4)*8
//   vector elems 0..7  = Bt[c][k0+kb .. k0+kb+7]
//   vector elems 8..15 = Bt[c][k0+kb+16 .. k0+kb+23]
__device__ __forceinline__ v16bf load_b_frag(const __bf16* __restrict__ Bt,
                                             int ldbt, int col0, int k0, int lane) {
    const int c  = lane & 15;
    const int kb = (lane >> 4) * 8;
    const __bf16* p = Bt + (size_t)(col0 + c) * (size_t)ldbt + (size_t)(k0 + kb);
    v8bf lo = *(const v8bf*)(p);
    v8bf hi = *(const v8bf*)(p + 16);
    v16bf b;
#pragma unroll
    for (int i = 0; i < 8; ++i) { b[i] = lo[i]; b[8 + i] = hi[i]; }
    return b;
}

// ---------------- GEMM: C[Mm x Nn] = scale * (A[Mm x Kk] @ Bt[Nn x Kk]^T) ---
// Block = 128 threads (4 waves). Block tile = 128 rows x 64 cols.
// Wave tile = 32 rows x 64 cols -> 8 accumulator fragments.
template <bool OUT_F32, bool OUT_TRANS>
__global__ __launch_bounds__(128) void gemm_bt_kernel(
    const __bf16* __restrict__ A, const __bf16* __restrict__ Bt,
    void* __restrict__ Cout, int Mm, int Nn, int Kk, float scale)
{
    const int lane = threadIdx.x & 31;
    const int wave = threadIdx.x >> 5;
    const int row0 = blockIdx.y * 128 + wave * 32;
    const int col0 = blockIdx.x * 64;

    const v8f zero = {0.f, 0.f, 0.f, 0.f, 0.f, 0.f, 0.f, 0.f};
    v8f acc[2][4];
#pragma unroll
    for (int mi = 0; mi < 2; ++mi)
#pragma unroll
        for (int t = 0; t < 4; ++t) acc[mi][t] = zero;

    for (int k0 = 0; k0 < Kk; k0 += 32) {
        v16bf a0 = load_a_frag(A, Kk, row0,      k0, lane);
        v16bf a1 = load_a_frag(A, Kk, row0 + 16, k0, lane);
#pragma unroll
        for (int t = 0; t < 4; ++t) {
            v16bf b = load_b_frag(Bt, Kk, col0 + 16 * t, k0, lane);
            acc[0][t] = __builtin_amdgcn_wmma_f32_16x16x32_bf16(
                false, a0, false, b, (short)0, acc[0][t], false, false);
            acc[1][t] = __builtin_amdgcn_wmma_f32_16x16x32_bf16(
                false, a1, false, b, (short)0, acc[1][t], false, false);
        }
    }

    // C/D layout: lane l, vgpr r -> row = 8*(l>>4)+r, col = l&15
#pragma unroll
    for (int mi = 0; mi < 2; ++mi) {
        const int rbase = row0 + mi * 16 + (lane >> 4) * 8;
#pragma unroll
        for (int t = 0; t < 4; ++t) {
            const int cc = col0 + 16 * t + (lane & 15);
#pragma unroll
            for (int r = 0; r < 8; ++r) {
                const float v = acc[mi][t][r] * scale;
                if constexpr (OUT_F32) {
                    ((float*)Cout)[(size_t)(rbase + r) * (size_t)Nn + cc] = v;
                } else if constexpr (OUT_TRANS) {
                    ((__bf16*)Cout)[(size_t)cc * (size_t)Mm + (rbase + r)] = (__bf16)v;
                } else {
                    ((__bf16*)Cout)[(size_t)(rbase + r) * (size_t)Nn + cc] = (__bf16)v;
                }
            }
        }
    }
}

// ---------------- prep kernels ---------------------------------------------

__global__ __launch_bounds__(256) void cast_bf16_kernel(
    const float* __restrict__ in, __bf16* __restrict__ out, int nElem)
{
    const int i = blockIdx.x * 256 + threadIdx.x;
    if (i < nElem) out[i] = (__bf16)in[i];
}

// out[c][r] = (bf16) in[r][c] ; in is [R x C] f32 row-major.
__global__ __launch_bounds__(256) void transpose_cast_kernel(
    const float* __restrict__ in, __bf16* __restrict__ out, int R, int C)
{
    __shared__ float tile[32][33];
    const int tx = threadIdx.x & 31;
    const int ty = threadIdx.x >> 5;            // 0..7
    const int c0 = blockIdx.x * 32;
    const int r0 = blockIdx.y * 32;
#pragma unroll
    for (int i = 0; i < 32; i += 8)
        tile[ty + i][tx] = in[(size_t)(r0 + ty + i) * (size_t)C + (c0 + tx)];
    __syncthreads();
#pragma unroll
    for (int i = 0; i < 32; i += 8)
        out[(size_t)(c0 + ty + i) * (size_t)R + (r0 + tx)] = (__bf16)tile[tx][ty + i];
}

// ---------------- softmax over rows of att[N_TOK x N_TOK] (bf16, in place) --

__global__ __launch_bounds__(256) void softmax_rows_kernel(__bf16* __restrict__ att)
{
    const int row  = blockIdx.x;
    const int tid  = threadIdx.x;
    const int lane = tid & 31;
    const int wv   = tid >> 5;
    __bf16* p = att + (size_t)row * (size_t)N_TOK;

    __shared__ float red[8];
    const int PER = N_TOK / 256;                // 16 elements per thread
    float e[N_TOK / 256];

    float m = -3.0e38f;
#pragma unroll
    for (int j = 0; j < PER; ++j) {
        e[j] = (float)p[tid + j * 256];
        m = fmaxf(m, e[j]);
    }
#pragma unroll
    for (int off = 16; off > 0; off >>= 1) m = fmaxf(m, __shfl_xor(m, off, 32));
    if (lane == 0) red[wv] = m;
    __syncthreads();
    float rowmax = red[0];
#pragma unroll
    for (int i = 1; i < 8; ++i) rowmax = fmaxf(rowmax, red[i]);
    __syncthreads();

    float s = 0.f;
#pragma unroll
    for (int j = 0; j < PER; ++j) { e[j] = __expf(e[j] - rowmax); s += e[j]; }
#pragma unroll
    for (int off = 16; off > 0; off >>= 1) s += __shfl_xor(s, off, 32);
    if (lane == 0) red[wv] = s;
    __syncthreads();
    float tot = 0.f;
#pragma unroll
    for (int i = 0; i < 8; ++i) tot += red[i];
    const float inv = 1.0f / tot;
#pragma unroll
    for (int j = 0; j < PER; ++j) p[tid + j * 256] = (__bf16)(e[j] * inv);
}

// ---------------- launcher --------------------------------------------------

extern "C" void kernel_launch(void* const* d_in, const int* in_sizes, int n_in,
                              void* d_out, int out_size, void* d_ws, size_t ws_size,
                              hipStream_t stream) {
    (void)in_sizes; (void)n_in; (void)out_size; (void)ws_size;
    const float* x  = (const float*)d_in[0];
    const float* Wk = (const float*)d_in[1];
    const float* Wq = (const float*)d_in[2];
    const float* Wv = (const float*)d_in[3];
    const float* Wo = (const float*)d_in[4];
    float* out = (float*)d_out;

    // Workspace layout (80 MB total)
    char* ws = (char*)d_ws;
    __bf16* x_bf  = (__bf16*)(ws);                        //  8 MB  [4096 x 1024]
    __bf16* WkT   = (__bf16*)(ws + (8u  << 20));          //  2 MB  [1024 x 1024]
    __bf16* WqT   = (__bf16*)(ws + (10u << 20));          //  2 MB
    __bf16* WvT   = (__bf16*)(ws + (12u << 20));          //  2 MB
    __bf16* WoT   = (__bf16*)(ws + (14u << 20));          //  2 MB
    __bf16* K_bf  = (__bf16*)(ws + (16u << 20));          //  8 MB  [4096 x 1024]
    __bf16* Q_bf  = (__bf16*)(ws + (24u << 20));          //  8 MB  [4096 x 1024]
    __bf16* Vt_bf = (__bf16*)(ws + (32u << 20));          //  8 MB  [1024 x 4096]
    __bf16* att   = (__bf16*)(ws + (40u << 20));          // 32 MB  [4096 x 4096]
    __bf16* y_bf  = (__bf16*)(ws + (72u << 20));          //  8 MB  [4096 x 1024]

    // 1) casts / weight transposes
    cast_bf16_kernel<<<(N_TOK * D_IN) / 256, 256, 0, stream>>>(x, x_bf, N_TOK * D_IN);
    dim3 tgrid(D_OUT / 32, D_IN / 32);
    transpose_cast_kernel<<<tgrid, 256, 0, stream>>>(Wk, WkT, D_IN, D_OUT);
    transpose_cast_kernel<<<tgrid, 256, 0, stream>>>(Wq, WqT, D_IN, D_OUT);
    transpose_cast_kernel<<<tgrid, 256, 0, stream>>>(Wv, WvT, D_IN, D_OUT);
    transpose_cast_kernel<<<tgrid, 256, 0, stream>>>(Wo, WoT, D_OUT, D_IN);

    // 2) projections: K = x Wk ; Q = 0.06 x Wq ; V^T = (x Wv)^T
    dim3 gproj(D_OUT / 64, N_TOK / 128);
    gemm_bt_kernel<false, false><<<gproj, 128, 0, stream>>>(
        x_bf, WkT, K_bf, N_TOK, D_OUT, D_IN, 1.0f);
    gemm_bt_kernel<false, false><<<gproj, 128, 0, stream>>>(
        x_bf, WqT, Q_bf, N_TOK, D_OUT, D_IN, Q_SCALE);
    gemm_bt_kernel<false, true><<<gproj, 128, 0, stream>>>(
        x_bf, WvT, Vt_bf, N_TOK, D_OUT, D_IN, 1.0f);

    // 3) logits = Q @ K^T  (Bt operand is K itself, row-major)
    dim3 glog(N_TOK / 64, N_TOK / 128);
    gemm_bt_kernel<false, false><<<glog, 128, 0, stream>>>(
        Q_bf, K_bf, att, N_TOK, N_TOK, D_OUT, 1.0f);

    // 4) row softmax in place
    softmax_rows_kernel<<<N_TOK, 256, 0, stream>>>(att);

    // 5) y = att @ V   (Bt operand = V^T)
    dim3 gav(D_OUT / 64, N_TOK / 128);
    gemm_bt_kernel<false, false><<<gav, 128, 0, stream>>>(
        att, Vt_bf, y_bf, N_TOK, D_OUT, N_TOK, 1.0f);

    // 6) out = y @ Wo  (f32 output)
    dim3 gout(D_IN / 64, N_TOK / 128);
    gemm_bt_kernel<true, false><<<gout, 128, 0, stream>>>(
        y_bf, WoT, (void*)out, N_TOK, D_IN, D_OUT, 1.0f);
}
